// MultiHeadAttention_4922032521487
// MI455X (gfx1250) — matrix operand order repaired and checked by reading, not compiled, not run on hardware
//
#include <hip/hip_runtime.h>

// ---------------------------------------------------------------------------
// MHA forward for MI455X (gfx1250, wave32, WMMA).
// B=4, N=2048, C=1024, H=16, D=64.  bf16 WMMA with fp32 accumulation.
// v2: 32x32 register-blocked GEMM tiles; block-cooperative flash attention
//     with async global->LDS staging of K/V tiles (ASYNCcnt path).
// ---------------------------------------------------------------------------

typedef unsigned short u16;
typedef __bf16 v16bf __attribute__((ext_vector_type(16)));
typedef float  v8f   __attribute__((ext_vector_type(8)));
typedef u16    u16x8 __attribute__((ext_vector_type(8)));

union FragBF {
    v16bf v;
    u16x8 h[2];
};

__device__ __forceinline__ u16 f2bf(float f) {
    unsigned u = __builtin_bit_cast(unsigned, f);
    unsigned r = u + 0x7FFFu + ((u >> 16) & 1u);   // round-to-nearest-even
    return (u16)(r >> 16);
}

__device__ __forceinline__ v8f wmma_bf16(const FragBF& a, const FragBF& b, v8f c) {
    return __builtin_amdgcn_wmma_f32_16x16x32_bf16(false, a.v, false, b.v,
                                                   (short)0, c, false, false);
}

__device__ __forceinline__ float rowmax16(float v) {
    for (int m = 8; m; m >>= 1) v = fmaxf(v, __shfl_xor(v, m, 16));
    return v;
}
__device__ __forceinline__ float rowsum16(float v) {
    for (int m = 8; m; m >>= 1) v += __shfl_xor(v, m, 16);
    return v;
}

// Async global -> LDS copy of 16 bytes (tracked by ASYNCcnt).
// VDST carries the per-lane LDS byte address; generic shared pointers map to
// LDS via addr[31:0] (ISA 10.2 aperture rules).
__device__ __forceinline__ void async_g2l_b128(const void* gptr, void* lptr) {
    unsigned lds_off = (unsigned)(size_t)lptr;
    unsigned long long ga = (unsigned long long)(size_t)gptr;
    asm volatile("global_load_async_to_lds_b128 %0, %1, off"
                 :: "v"(lds_off), "v"(ga) : "memory");
}
__device__ __forceinline__ void wait_async0() {
    asm volatile("s_wait_asynccnt 0" ::: "memory");
}

// ---------------------------------------------------------------------------
// fp32 -> bf16 conversion kernels
// ---------------------------------------------------------------------------
__global__ void f32_to_bf16_kernel(const float* __restrict__ in,
                                   u16* __restrict__ out, int n) {
    int i = blockIdx.x * blockDim.x + threadIdx.x;
    int stride = gridDim.x * blockDim.x;
    for (; i < n; i += stride) out[i] = f2bf(in[i]);
}

// in: [rows, cols] fp32 -> out: [cols, rows] bf16 (K-major weights)
__global__ void transpose_f32_to_bf16_kernel(const float* __restrict__ in,
                                             u16* __restrict__ out,
                                             int rows, int cols) {
    int i = blockIdx.x * blockDim.x + threadIdx.x;
    int stride = gridDim.x * blockDim.x;
    int n = rows * cols;
    for (; i < n; i += stride) {
        int j = i / rows;
        int k = i % rows;
        out[i] = f2bf(in[(size_t)k * cols + j]);
    }
}

// ---------------------------------------------------------------------------
// 32x32 output tile per wave: 2 A-frags x 2 B-frags -> 4 WMMAs per k-step.
// A: [M, K] row-major bf16.  Bt: [Ncols, K] row-major bf16 (i.e. B^T).
// ---------------------------------------------------------------------------
__device__ __forceinline__ void gemm_tile32(const u16* __restrict__ A,
                                            const u16* __restrict__ Bt,
                                            int lda, int ldb, int K,
                                            int row0, int col0,
                                            v8f (&acc)[2][2]) {
    const int lane = threadIdx.x & 31;
    const int hi = lane >> 4, lm = lane & 15;
    const u16* ap0 = A  + (size_t)(row0 + lm) * lda + 8 * hi;
    const u16* ap1 = ap0 + (size_t)16 * lda;
    const u16* bp0 = Bt + (size_t)(col0 + lm) * ldb + 8 * hi;
    const u16* bp1 = bp0 + (size_t)16 * ldb;
    for (int kk = 0; kk < K; kk += 32) {
        FragBF a0, a1, b0, b1;
        a0.h[0] = *(const u16x8*)(ap0);
        a0.h[1] = *(const u16x8*)(ap0 + 16);
        a1.h[0] = *(const u16x8*)(ap1);
        a1.h[1] = *(const u16x8*)(ap1 + 16);
        b0.h[0] = *(const u16x8*)(bp0);
        b0.h[1] = *(const u16x8*)(bp0 + 16);
        b1.h[0] = *(const u16x8*)(bp1);
        b1.h[1] = *(const u16x8*)(bp1 + 16);
        ap0 += 32; ap1 += 32; bp0 += 32; bp1 += 32;
        __builtin_prefetch(ap0 + 32, 0, 1);
        __builtin_prefetch(bp0 + 32, 0, 1);
        acc[0][0] = wmma_bf16(a0, b0, acc[0][0]);
        acc[0][1] = wmma_bf16(a0, b1, acc[0][1]);
        acc[1][0] = wmma_bf16(a1, b0, acc[1][0]);
        acc[1][1] = wmma_bf16(a1, b1, acc[1][1]);
    }
}

// ---------------------------------------------------------------------------
// QKV GEMM: [8192,1024] x [1024,3072] + bias -> Q,K [B,H,N,D], V^T [B,H,D,N]
// grid (128, 24), block 256: 8 waves as 2x4 of 32x32 tiles = 64x128 per block
// ---------------------------------------------------------------------------
__global__ void qkv_gemm_kernel(const u16* __restrict__ xb,
                                const u16* __restrict__ wT,
                                const float* __restrict__ bias,
                                u16* __restrict__ Q,
                                u16* __restrict__ Kb,
                                u16* __restrict__ Vt) {
    const int wave = threadIdx.x >> 5;
    const int row0 = blockIdx.x * 64 + (wave >> 2) * 32;
    const int col0 = blockIdx.y * 128 + (wave & 3) * 32;
    v8f acc[2][2] = {};
    gemm_tile32(xb, wT, 1024, 1024, 1024, row0, col0, acc);

    const int lane = threadIdx.x & 31;
    const int hi = lane >> 4, lm = lane & 15;
#pragma unroll
    for (int ca = 0; ca < 2; ++ca) {
        const int col = col0 + ca * 16 + lm;
        const float bc = bias[col];
        const int which = col >> 10;       // 0=q, 1=k, 2=v
        const int rem = col & 1023;
        const int h = rem >> 6, d = rem & 63;
#pragma unroll
        for (int ra = 0; ra < 2; ++ra) {
#pragma unroll
            for (int r = 0; r < 8; ++r) {
                int rowi = row0 + ra * 16 + r + 8 * hi;   // [0, 8192)
                int b = rowi >> 11, n = rowi & 2047;
                u16 bf = f2bf(acc[ra][ca][r] + bc);
                size_t bh = (size_t)(b * 16 + h);
                if (which == 0)      Q [(bh * 2048 + n) * 64 + d] = bf;
                else if (which == 1) Kb[(bh * 2048 + n) * 64 + d] = bf;
                else                 Vt[(bh * 64 + d) * 2048 + n] = bf;
            }
        }
    }
}

// ---------------------------------------------------------------------------
// Output projection: attn[8192,1024] x [1024,1024] + bias -> fp32 out
// grid (128, 8), block 256
// ---------------------------------------------------------------------------
__global__ void proj_gemm_kernel(const u16* __restrict__ attn,
                                 const u16* __restrict__ wT,
                                 const float* __restrict__ bias,
                                 float* __restrict__ out) {
    const int wave = threadIdx.x >> 5;
    const int row0 = blockIdx.x * 64 + (wave >> 2) * 32;
    const int col0 = blockIdx.y * 128 + (wave & 3) * 32;
    v8f acc[2][2] = {};
    gemm_tile32(attn, wT, 1024, 1024, 1024, row0, col0, acc);

    const int lane = threadIdx.x & 31;
    const int hi = lane >> 4, lm = lane & 15;
#pragma unroll
    for (int ca = 0; ca < 2; ++ca) {
        const int col = col0 + ca * 16 + lm;
        const float bc = bias[col];
#pragma unroll
        for (int ra = 0; ra < 2; ++ra) {
#pragma unroll
            for (int r = 0; r < 8; ++r) {
                int rowi = row0 + ra * 16 + r + 8 * hi;
                out[(size_t)rowi * 1024 + col] = acc[ra][ca][r] + bc;
            }
        }
    }
}

// ---------------------------------------------------------------------------
// Block-cooperative causal flash attention.
// Block = 8 waves = 128 query rows of one (b,h); K/V tiles for 32 keys are
// staged once per block into LDS via async global->LDS loads (ASYNCcnt),
// then every wave reads its WMMA fragments from LDS (ds_load_b128).
// Q,K: [B*H, N, D] bf16.  Vt: [B*H, D, N] bf16.  attn out: [B, N, H*D] bf16.
// grid 1024, block 256.
// ---------------------------------------------------------------------------
__global__ void flash_attn_kernel(const u16* __restrict__ Q,
                                  const u16* __restrict__ K,
                                  const u16* __restrict__ Vt,
                                  u16* __restrict__ attn) {
    __shared__ alignas(16) u16 kbuf[32 * 64];   // [key_local][d]      4 KB
    __shared__ alignas(16) u16 vbuf[64 * 32];   // [d][key_local]      4 KB
    __shared__ alignas(16) u16 pbuf[8][512];    // per-wave 16x32 P    8 KB

    const int tid  = threadIdx.x;
    const int wave = tid >> 5;
    const int lane = tid & 31;
    const int hi = lane >> 4, lm = lane & 15;

    const int bh = blockIdx.x >> 4;             // (b*16 + h)
    const int Q0 = (blockIdx.x & 15) * 128;     // block's first query row
    const int q0 = Q0 + wave * 16;              // wave's first query row

    const u16* Qp = Q  + (size_t)bh * 2048 * 64;
    const u16* Kp = K  + (size_t)bh * 2048 * 64;
    const u16* Vp = Vt + (size_t)bh * 64 * 2048;

    // Q A-fragments for d = [0,32) and [32,64)
    FragBF qa[2];
    {
        const u16* qp = Qp + (size_t)(q0 + lm) * 64 + 8 * hi;
        qa[0].h[0] = *(const u16x8*)(qp);
        qa[0].h[1] = *(const u16x8*)(qp + 16);
        qa[1].h[0] = *(const u16x8*)(qp + 32);
        qa[1].h[1] = *(const u16x8*)(qp + 48);
    }

    v8f acc0 = {}, acc1 = {}, acc2 = {}, acc3 = {};
    float mrun[8], lrun[8];
#pragma unroll
    for (int r = 0; r < 8; ++r) { mrun[r] = -1e30f; lrun[r] = 0.0f; }

    const float SCALE = 0.125f;                 // 64^-0.5
    const int kv_d = tid >> 2;                  // V copy: d row
    const int kv_c = tid & 3;                   // V copy: 16B chunk in row

    for (int jj = 0; jj <= Q0 + 127; jj += 32) {
        // ---- stage K (32x64) and V^T (64x32) tiles: 16 B per thread each ----
        // K rows jj..jj+31 are one contiguous 4 KB block.
        async_g2l_b128(Kp + (size_t)jj * 64 + tid * 8, &kbuf[tid * 8]);
        // V^T row d covers keys jj..jj+31 (64 B, contiguous).
        async_g2l_b128(Vp + (size_t)kv_d * 2048 + jj + kv_c * 8,
                       &vbuf[kv_d * 32 + kv_c * 8]);
        wait_async0();
        __syncthreads();

        // ---- S = Q * K^T for keys [jj, jj+32): two 16x16 f32 tiles ----
        v8f s0 = {}, s1 = {};
#pragma unroll
        for (int ks = 0; ks < 2; ++ks) {
            FragBF kb0, kb1;
            const u16* kp0 = &kbuf[lm * 64 + ks * 32 + 8 * hi];
            const u16* kp1 = kp0 + 16 * 64;
            kb0.h[0] = *(const u16x8*)(kp0);
            kb0.h[1] = *(const u16x8*)(kp0 + 16);
            kb1.h[0] = *(const u16x8*)(kp1);
            kb1.h[1] = *(const u16x8*)(kp1 + 16);
            s0 = wmma_bf16(qa[ks], kb0, s0);
            s1 = wmma_bf16(qa[ks], kb1, s1);
        }

        // ---- online softmax update (rows span half-wave lanes) ----
#pragma unroll
        for (int r = 0; r < 8; ++r) {
            int qrow = q0 + r + 8 * hi;
            float v0 = s0[r] * SCALE;
            float v1 = s1[r] * SCALE;
            if (jj + lm > qrow)      v0 = -1e30f;   // causal mask
            if (jj + 16 + lm > qrow) v1 = -1e30f;
            float rm = rowmax16(fmaxf(v0, v1));
            float mt = fmaxf(mrun[r], rm);
            float p0 = __expf(v0 - mt);
            float p1 = __expf(v1 - mt);
            float rs = rowsum16(p0 + p1);
            float alpha = __expf(mrun[r] - mt);
            mrun[r] = mt;
            lrun[r] = lrun[r] * alpha + rs;
            acc0[r] *= alpha; acc1[r] *= alpha;
            acc2[r] *= alpha; acc3[r] *= alpha;
            int rr = r + 8 * hi;
            pbuf[wave][rr * 32 + lm]      = f2bf(p0);
            pbuf[wave][rr * 32 + 16 + lm] = f2bf(p1);
        }
        asm volatile("s_wait_dscnt 0" ::: "memory");   // per-wave P round-trip

        // ---- P as A-fragment (16 queries x 32 keys) from LDS ----
        FragBF pa;
        {
            const u16* pp = &pbuf[wave][lm * 32 + 8 * hi];
            pa.h[0] = *(const u16x8*)(pp);
            pa.h[1] = *(const u16x8*)(pp + 16);
        }

        // ---- O += P * V  (V^T tile: columns contiguous over keys) ----
#pragma unroll
        for (int dt = 0; dt < 4; ++dt) {
            FragBF vb;
            const u16* vp = &vbuf[(dt * 16 + lm) * 32 + 8 * hi];
            vb.h[0] = *(const u16x8*)(vp);
            vb.h[1] = *(const u16x8*)(vp + 16);
            if (dt == 0)      acc0 = wmma_bf16(pa, vb, acc0);
            else if (dt == 1) acc1 = wmma_bf16(pa, vb, acc1);
            else if (dt == 2) acc2 = wmma_bf16(pa, vb, acc2);
            else              acc3 = wmma_bf16(pa, vb, acc3);
        }
        __syncthreads();   // all waves done with kbuf/vbuf before restage
    }

    // ---- epilogue: normalize and store bf16 [B, N, H*D] ----
    const int h = bh & 15, b = bh >> 4;
#pragma unroll
    for (int r = 0; r < 8; ++r) {
        int n = q0 + r + 8 * hi;
        float inv = 1.0f / lrun[r];
        size_t base = ((size_t)(b * 2048 + n)) * 1024 + h * 64 + lm;
        attn[base + 0]  = f2bf(acc0[r] * inv);
        attn[base + 16] = f2bf(acc1[r] * inv);
        attn[base + 32] = f2bf(acc2[r] * inv);
        attn[base + 48] = f2bf(acc3[r] * inv);
    }
}

// ---------------------------------------------------------------------------
// Host launch
// ---------------------------------------------------------------------------
extern "C" void kernel_launch(void* const* d_in, const int* in_sizes, int n_in,
                              void* d_out, int out_size, void* d_ws, size_t ws_size,
                              hipStream_t stream) {
    (void)in_sizes; (void)n_in; (void)out_size; (void)ws_size;
    const float* x      = (const float*)d_in[0];   // [4,2048,1024]
    const float* w_qkv  = (const float*)d_in[1];   // [1024,3072]
    const float* b_qkv  = (const float*)d_in[2];   // [3072]
    const float* w_proj = (const float*)d_in[3];   // [1024,1024]
    const float* b_proj = (const float*)d_in[4];   // [1024]
    // d_in[5] = causal mask, applied analytically in the kernel
    float* out = (float*)d_out;

    char* ws = (char*)d_ws;
    size_t off = 0;
    auto carve = [&](size_t bytes) -> void* {
        void* p = ws + off;
        off += (bytes + 255) & ~(size_t)255;
        return p;
    };
    u16* xb   = (u16*)carve((size_t)8192 * 1024 * 2);
    u16* wqT  = (u16*)carve((size_t)3072 * 1024 * 2);
    u16* wpT  = (u16*)carve((size_t)1024 * 1024 * 2);
    u16* Qb   = (u16*)carve((size_t)64 * 2048 * 64 * 2);
    u16* Kb   = (u16*)carve((size_t)64 * 2048 * 64 * 2);
    u16* Vt   = (u16*)carve((size_t)64 * 64 * 2048 * 2);
    u16* attn = (u16*)carve((size_t)8192 * 1024 * 2);

    f32_to_bf16_kernel<<<4096, 256, 0, stream>>>(x, xb, 8192 * 1024);
    transpose_f32_to_bf16_kernel<<<4096, 256, 0, stream>>>(w_qkv, wqT, 1024, 3072);
    transpose_f32_to_bf16_kernel<<<2048, 256, 0, stream>>>(w_proj, wpT, 1024, 1024);

    qkv_gemm_kernel<<<dim3(128, 24), 256, 0, stream>>>(xb, wqT, b_qkv, Qb, Kb, Vt);
    flash_attn_kernel<<<1024, 256, 0, stream>>>(Qb, Kb, Vt, attn);
    proj_gemm_kernel<<<dim3(128, 8), 256, 0, stream>>>(attn, wpT, b_proj, out);
}
